// CPGPT2_42717744726825
// MI455X (gfx1250) — compile-verified, matrix operand order checked by reading, not run on hardware
//
#include <hip/hip_runtime.h>
#include <hip/hip_bf16.h>
#include <math.h>

// Problem constants (must match reference)
#define B_   2
#define T_   256
#define V_   512
#define E_   768
#define L_   12
#define NH_  12
#define HD_  64
#define HCP_ 2
#define R_   2
#define M_   (B_*T_)          // 512 rows of activations
#define NVALID_ (B_*(T_-HCP_))// 508 tokens contributing to loss

typedef __attribute__((ext_vector_type(16))) _Float16 v16h;
typedef __attribute__((ext_vector_type(8)))  _Float16 v8h;
typedef __attribute__((ext_vector_type(8)))  float    v8f;
typedef unsigned int u32x4 __attribute__((ext_vector_type(4)));
typedef int          i32x4 __attribute__((ext_vector_type(4)));
typedef int          i32x8 __attribute__((ext_vector_type(8)));

__device__ __forceinline__ float gelu_f(float x) {
    float x3 = x*x*x;
    return 0.5f*x*(1.0f + tanhf(0.7978845608028654f*(x + 0.044715f*x3)));
}

// ---------------------------------------------------------------------------
// Tensor Data Mover: 2D f32 tile (tile_x elems per row, tile_y rows, row
// stride in elements) from global -> contiguous LDS. One issue per wave.
// D# layout per CDNA5 ISA. 6-arg builtin form
// (clang-23): (g0 u32x4, g1 i32x8, g2 i32x4, g3 i32x4, i32x8, cpol).
// ---------------------------------------------------------------------------
__device__ __forceinline__ void tdm_load_tile_f32(unsigned lds_off,
        const float* gptr, unsigned tile_x, unsigned tile_y,
        unsigned long long row_stride_elems) {
    unsigned long long ga = (unsigned long long)(uintptr_t)gptr;
    u32x4 g0;
    g0[0] = 1u;                                          // count=1 (valid user D#)
    g0[1] = lds_off;                                     // lds_addr (bytes)
    g0[2] = (unsigned)ga;                                // global_addr[31:0]
    g0[3] = (unsigned)((ga >> 32) & 0x01FFFFFFull)       // global_addr[56:32]
          | (2u << 30);                                  // type=2 ("image")
    i32x8 g1;
    g1[0] = (int)(2u << 16);                             // wg_mask=0, data_size=2 (4B)
    g1[1] = (int)((tile_x & 0xFFFFu) << 16);             // tensor_dim0[15:0]
    g1[2] = (int)((tile_y & 0xFFFFu) << 16);             // tensor_dim0[31:16]=0 | tensor_dim1[15:0]
    g1[3] = (int)((tile_x & 0xFFFFu) << 16);             // tensor_dim1[31:16]=0 | tile_dim0
    g1[4] = (int)(tile_y & 0xFFFFu);                     // tile_dim1 | tile_dim2=0
    g1[5] = (int)(unsigned)(row_stride_elems & 0xFFFFFFFFull);      // dim0_stride[31:0]
    g1[6] = (int)(unsigned)((row_stride_elems >> 32) & 0xFFFFull);  // dim0_stride[47:32] | dim1_stride[15:0]=0
    g1[7] = 0;
    i32x4 z4 = {0,0,0,0};
    i32x8 z8 = {0,0,0,0,0,0,0,0};
    __builtin_amdgcn_tensor_load_to_lds(g0, g1, z4, z4, z8, 0);
}

// ---------------------------------------------------------------------------
// Embedding: h[b,t,:] = wte[ids[b,t],:] + wpe[:T][t,:]
// ---------------------------------------------------------------------------
__global__ void k_embed(const int* __restrict__ ids, const float* __restrict__ wte,
                        const float* __restrict__ wpe, float* __restrict__ h) {
    int row = blockIdx.x;            // 0..M_-1
    int t   = row % T_;
    int id  = ids[row];
    for (int c = threadIdx.x; c < E_; c += blockDim.x)
        h[(size_t)row*E_ + c] = wte[(size_t)id*E_ + c] + wpe[(size_t)t*E_ + c];
}

// ---------------------------------------------------------------------------
// Tiled cast+transpose: f32 (K,N) row-major -> f16 (N,K) row-major.
// 32x32 LDS tile -> coalesced reads AND writes. All dims are multiples of 32.
// Pre-transposed f16 weights make both WMMA fragments contiguous b128 loads,
// and the full f16 weight set (~170MB) fits in the 192MB L2.
// ---------------------------------------------------------------------------
__global__ void k_castT(const float* __restrict__ src, _Float16* __restrict__ dst,
                        int K, int N) {
    __shared__ _Float16 tile[32][33];
    const int kt = blockIdx.y*32, nt = blockIdx.x*32;
    const int tx = threadIdx.x & 31, ty = threadIdx.x >> 5;   // 256 thr = 8 rows/pass
    #pragma unroll
    for (int r = ty; r < 32; r += 8)
        tile[r][tx] = (_Float16)src[(size_t)(kt + r)*N + (nt + tx)];
    __syncthreads();
    #pragma unroll
    for (int r = ty; r < 32; r += 8)
        dst[(size_t)(nt + r)*K + (kt + tx)] = tile[tx][r];
}

// ---------------------------------------------------------------------------
// LayerNorm over E_=768, one block per row, output f16 (GEMM A operand)
// ---------------------------------------------------------------------------
__global__ void k_layernorm(const float* __restrict__ in, const float* __restrict__ g,
                            const float* __restrict__ b, _Float16* __restrict__ out) {
    int row = blockIdx.x;
    const float* x = in + (size_t)row*E_;
    __shared__ float red[256];
    float s = 0.f;
    for (int c = threadIdx.x; c < E_; c += 256) s += x[c];
    red[threadIdx.x] = s; __syncthreads();
    for (int o = 128; o > 0; o >>= 1) {
        if (threadIdx.x < o) red[threadIdx.x] += red[threadIdx.x+o];
        __syncthreads();
    }
    float mean = red[0] * (1.f/E_);
    __syncthreads();
    s = 0.f;
    for (int c = threadIdx.x; c < E_; c += 256) { float d = x[c]-mean; s += d*d; }
    red[threadIdx.x] = s; __syncthreads();
    for (int o = 128; o > 0; o >>= 1) {
        if (threadIdx.x < o) red[threadIdx.x] += red[threadIdx.x+o];
        __syncthreads();
    }
    float rstd = rsqrtf(red[0]*(1.f/E_) + 1e-5f);
    for (int c = threadIdx.x; c < E_; c += 256)
        out[(size_t)row*E_ + c] = (_Float16)((x[c]-mean)*rstd*g[c] + b[c]);
}

// ---------------------------------------------------------------------------
// WMMA GEMM: C(M,N) = A(M,K) @ WT(N,K)^T + bias, epilogue variants.
// One wave per 16x64 output tile (A fragment reused across 4 WMMAs), K-loop
// software-pipelined with register double-buffering: fragment loads for step
// k+32 are issued before the WMMAs of step k consume the current fragments,
// so the scheduler can overlap the 10 in-flight b128 loads with matrix issue.
// Fragment layout per CDNA5 ISA 7.12.2 (16-bit A 16x32): lanes 0-15 hold
// K=[0..7]+[16..23], lanes 16-31 hold K=[8..15]+[24..31] for row M=lane&15;
// B mirrors it with column N=lane&15 (pre-transposed weights -> 16B loads).
// EPI: 0 = store f32 (acc+bias); 1 = residual: f32 res + acc + bias;
//      2 = gelu(acc+bias) stored as f16 (feeds next GEMM A operand)
// ---------------------------------------------------------------------------
__device__ __forceinline__ v16h load_frag(const _Float16* p, int k0) {
    v8h lo = *(const v8h*)(p + k0);
    v8h hi = *(const v8h*)(p + k0 + 16);
    return __builtin_shufflevector(lo, hi, 0,1,2,3,4,5,6,7,8,9,10,11,12,13,14,15);
}

template<int EPI>
__global__ void k_gemm_wmma(const _Float16* __restrict__ A, const _Float16* __restrict__ WT,
                            const float* __restrict__ bias, const float* __restrict__ res,
                            float* __restrict__ outF, _Float16* __restrict__ outH,
                            int M, int N, int K) {
    const int lane  = threadIdx.x & 31;
    const int half  = lane >> 4;   // 0/1: selects K sub-range for A/B frags
    const int l15   = lane & 15;
    const int tileM = blockIdx.y * 16;
    const int tileN = blockIdx.x * 64;

    const _Float16* Ap = A + (size_t)(tileM + l15)*K + half*8;
    const _Float16* Bp[4];
    #pragma unroll
    for (int tn = 0; tn < 4; ++tn)
        Bp[tn] = WT + (size_t)(tileN + tn*16 + l15)*K + half*8;

    v8f acc[4];
    #pragma unroll
    for (int tn = 0; tn < 4; ++tn) acc[tn] = (v8f){0.f,0.f,0.f,0.f,0.f,0.f,0.f,0.f};

    // prologue: fragments for k0 = 0
    v16h aF = load_frag(Ap, 0);
    v16h bF[4];
    #pragma unroll
    for (int tn = 0; tn < 4; ++tn) bF[tn] = load_frag(Bp[tn], 0);

    int k0 = 0;
    for (; k0 < K - 32; k0 += 32) {
        // issue next K-step loads first (double buffer)
        v16h aN = load_frag(Ap, k0 + 32);
        v16h bN[4];
        #pragma unroll
        for (int tn = 0; tn < 4; ++tn) bN[tn] = load_frag(Bp[tn], k0 + 32);
        // consume current fragments
        #pragma unroll
        for (int tn = 0; tn < 4; ++tn)
            acc[tn] = __builtin_amdgcn_wmma_f32_16x16x32_f16(
                          false, aF, false, bF[tn], (short)0, acc[tn], false, false);
        aF = aN;
        #pragma unroll
        for (int tn = 0; tn < 4; ++tn) bF[tn] = bN[tn];
    }
    // epilogue K-step
    #pragma unroll
    for (int tn = 0; tn < 4; ++tn)
        acc[tn] = __builtin_amdgcn_wmma_f32_16x16x32_f16(
                      false, aF, false, bF[tn], (short)0, acc[tn], false, false);

    // C/D layout: VGPR v holds row M = tileM + v + half*8, col N = tileN + l15
    const int mbase = tileM + half*8;
    #pragma unroll
    for (int tn = 0; tn < 4; ++tn) {
        const int ncol = tileN + tn*16 + l15;
        const float bv = bias[ncol];
        #pragma unroll
        for (int v = 0; v < 8; ++v) {
            size_t oidx = (size_t)(mbase + v)*N + ncol;
            float val = acc[tn][v] + bv;
            if (EPI == 1) val += res[oidx];
            if (EPI == 2) outH[oidx] = (_Float16)gelu_f(val);
            else          outF[oidx] = val;
        }
    }
}

// ---------------------------------------------------------------------------
// Attention: one block per (b,h); K/V 64x64 f32 tiles staged into LDS by the
// Tensor Data Mover (wave 0 issues two D# loads, waits TENSORcnt, barrier);
// per-thread online softmax over causal keys; output written as f16.
// ---------------------------------------------------------------------------
__global__ void k_attn(const float* __restrict__ qkv, _Float16* __restrict__ o16) {
    const int h = blockIdx.x, b = blockIdx.y;
    const int t = threadIdx.x;            // query row, 0..255
    __shared__ float sk[64][HD_];
    __shared__ float sv[64][HD_];

    const float* qrow = qkv + (size_t)(b*T_ + t)*3*E_ + h*HD_;
    float q[HD_], acc[HD_];
    #pragma unroll
    for (int d = 0; d < HD_; ++d) { q[d] = qrow[d]; acc[d] = 0.f; }
    float m = -1e30f, l = 0.f;
    const float scale = 0.125f;           // 1/sqrt(64)

    for (int c0 = 0; c0 < T_; c0 += 64) {
        __syncthreads();                  // previous chunk fully consumed
        if (threadIdx.x < 32) {           // wave 0 drives the TDM
            unsigned sk_off = (unsigned)(uintptr_t)(void*)&sk[0][0];
            unsigned sv_off = (unsigned)(uintptr_t)(void*)&sv[0][0];
            const float* kb = qkv + (size_t)(b*T_ + c0)*3*E_ +   E_ + h*HD_;
            const float* vb = qkv + (size_t)(b*T_ + c0)*3*E_ + 2*E_ + h*HD_;
            tdm_load_tile_f32(sk_off, kb, HD_, 64, (unsigned long long)(3*E_));
            tdm_load_tile_f32(sv_off, vb, HD_, 64, (unsigned long long)(3*E_));
            __builtin_amdgcn_s_wait_tensorcnt(0);
        }
        __syncthreads();                  // LDS tiles visible to all waves
        int smax = t - c0; if (smax > 63) smax = 63;
        for (int r = 0; r <= smax; ++r) {
            float x = 0.f;
            #pragma unroll
            for (int d = 0; d < HD_; ++d) x += q[d]*sk[r][d];
            x *= scale;
            float mnew = fmaxf(m, x);
            float corr = __expf(m - mnew);
            float p    = __expf(x - mnew);
            l = l*corr + p;
            #pragma unroll
            for (int d = 0; d < HD_; ++d) acc[d] = acc[d]*corr + p*sv[r][d];
            m = mnew;
        }
    }
    float inv = 1.f/l;
    _Float16* orow = o16 + (size_t)(b*T_ + t)*E_ + h*HD_;
    #pragma unroll
    for (int d = 0; d < HD_; ++d) orow[d] = (_Float16)(acc[d]*inv);
}

// ---------------------------------------------------------------------------
// CP loss: one block per valid token n. scores[i,j] = cp0[i]·cp1[j] (rank 2).
// Z = sum_ij exp(scores); loss_n = log Z - scores[t0,t1].
// ---------------------------------------------------------------------------
__global__ void k_cploss(const float* __restrict__ cp, const int* __restrict__ ids,
                         float* __restrict__ losses) {
    const int n = blockIdx.x;             // 0..NVALID_-1
    const int b = n / (T_ - HCP_);
    const int t = n % (T_ - HCP_);
    const int row = b*T_ + t;
    const float* c0 = cp + (size_t)row*(HCP_*V_*R_);
    const float* c1 = c0 + V_*R_;
    __shared__ float s0[V_*R_];
    __shared__ float s1[V_*R_];
    for (int i = threadIdx.x; i < V_*R_; i += 256) { s0[i] = c0[i]; s1[i] = c1[i]; }
    __syncthreads();

    float sum = 0.f;
    for (int i = 0; i < V_; ++i) {
        float a0 = s0[2*i], a1 = s0[2*i+1];
        for (int j = threadIdx.x; j < V_; j += 256)
            sum += __expf(a0*s1[2*j] + a1*s1[2*j+1]);
    }
    __shared__ float red[256];
    red[threadIdx.x] = sum; __syncthreads();
    for (int o = 128; o > 0; o >>= 1) {
        if (threadIdx.x < o) red[threadIdx.x] += red[threadIdx.x+o];
        __syncthreads();
    }
    if (threadIdx.x == 0) {
        float Z = red[0];
        int t0 = ids[b*T_ + t + 1];
        int t1 = ids[b*T_ + t + 2];
        float st = s0[2*t0]*s1[2*t1] + s0[2*t0+1]*s1[2*t1+1];
        losses[n] = logf(Z) - st;
    }
}

__global__ void k_reduce_loss(const float* __restrict__ losses, float* __restrict__ out) {
    __shared__ float red[256];
    float s = 0.f;
    for (int i = threadIdx.x; i < NVALID_; i += 256) s += losses[i];
    red[threadIdx.x] = s; __syncthreads();
    for (int o = 128; o > 0; o >>= 1) {
        if (threadIdx.x < o) red[threadIdx.x] += red[threadIdx.x+o];
        __syncthreads();
    }
    if (threadIdx.x == 0) out[0] = red[0] * (1.f/(float)NVALID_);
}

// ---------------------------------------------------------------------------
// Host side
// ---------------------------------------------------------------------------
extern "C" void kernel_launch(void* const* d_in, const int* in_sizes, int n_in,
                              void* d_out, int out_size, void* d_ws, size_t ws_size,
                              hipStream_t stream) {
    (void)in_sizes; (void)n_in; (void)out_size; (void)ws_size;
    const int*   ids   = (const int*)  d_in[0];
    const float* wte   = (const float*)d_in[1];
    const float* wpe   = (const float*)d_in[2];
    const float* ln1_g = (const float*)d_in[3];
    const float* ln1_b = (const float*)d_in[4];
    const float* ln2_g = (const float*)d_in[5];
    const float* ln2_b = (const float*)d_in[6];
    const float* w_qkv = (const float*)d_in[7];
    const float* b_qkv = (const float*)d_in[8];
    const float* w_ao  = (const float*)d_in[9];
    const float* b_ao  = (const float*)d_in[10];
    const float* w_fc  = (const float*)d_in[11];
    const float* b_fc  = (const float*)d_in[12];
    const float* w_mo  = (const float*)d_in[13];
    const float* b_mo  = (const float*)d_in[14];
    const float* lnf_g = (const float*)d_in[15];
    const float* lnf_b = (const float*)d_in[16];
    const float* w_cp  = (const float*)d_in[17];
    const float* b_cp  = (const float*)d_in[18];
    float* out = (float*)d_out;

    // Workspace carve-up (256B aligned)
    size_t off = 0;
    auto alloc = [&](size_t bytes) -> void* {
        off = (off + 255) & ~(size_t)255;
        void* p = (char*)d_ws + off;
        off += bytes;
        return p;
    };
    _Float16* wqT    = (_Float16*)alloc((size_t)L_*3*E_*E_*2);
    _Float16* waT    = (_Float16*)alloc((size_t)L_*E_*E_*2);
    _Float16* wfT    = (_Float16*)alloc((size_t)L_*4*E_*E_*2);
    _Float16* wmT    = (_Float16*)alloc((size_t)L_*4*E_*E_*2);
    _Float16* wcpT   = (_Float16*)alloc((size_t)E_*HCP_*V_*R_*2);
    float*    hbuf   = (float*)   alloc((size_t)M_*E_*4);
    _Float16* x16    = (_Float16*)alloc((size_t)M_*E_*2);
    float*    qkvb   = (float*)   alloc((size_t)M_*3*E_*4);
    _Float16* o16    = (_Float16*)alloc((size_t)M_*E_*2);
    _Float16* fc16   = (_Float16*)alloc((size_t)M_*4*E_*2);
    float*    cpbuf  = (float*)   alloc((size_t)M_*HCP_*V_*R_*4);
    float*    losses = (float*)   alloc((size_t)NVALID_*4);

    // --- weight prep: f32 -> transposed f16 (tiled; L2-resident afterwards)
    for (int l = 0; l < L_; ++l) {
        size_t nqkv = (size_t)E_*3*E_, nao = (size_t)E_*E_,
               nfc  = (size_t)E_*4*E_, nmo = (size_t)4*E_*E_;
        k_castT<<<dim3(3*E_/32,   E_/32), 256, 0, stream>>>(w_qkv + l*nqkv, wqT + l*nqkv, E_,   3*E_);
        k_castT<<<dim3(  E_/32,   E_/32), 256, 0, stream>>>(w_ao  + l*nao , waT + l*nao , E_,   E_);
        k_castT<<<dim3(4*E_/32,   E_/32), 256, 0, stream>>>(w_fc  + l*nfc , wfT + l*nfc , E_,   4*E_);
        k_castT<<<dim3(  E_/32, 4*E_/32), 256, 0, stream>>>(w_mo  + l*nmo , wmT + l*nmo , 4*E_, E_);
    }
    k_castT<<<dim3(HCP_*V_*R_/32, E_/32), 256, 0, stream>>>(w_cp, wcpT, E_, HCP_*V_*R_);

    // --- embedding
    k_embed<<<M_, 256, 0, stream>>>(ids, wte, wpe, hbuf);

    // --- transformer blocks
    for (int l = 0; l < L_; ++l) {
        k_layernorm<<<M_, 256, 0, stream>>>(hbuf, ln1_g + (size_t)l*E_, ln1_b + (size_t)l*E_, x16);

        dim3 gq(3*E_/64, M_/16);
        k_gemm_wmma<0><<<gq, 32, 0, stream>>>(x16, wqT + (size_t)l*E_*3*E_,
                                              b_qkv + (size_t)l*3*E_, nullptr,
                                              qkvb, nullptr, M_, 3*E_, E_);

        k_attn<<<dim3(NH_, B_), 256, 0, stream>>>(qkvb, o16);

        dim3 ga(E_/64, M_/16);
        k_gemm_wmma<1><<<ga, 32, 0, stream>>>(o16, waT + (size_t)l*E_*E_,
                                              b_ao + (size_t)l*E_, hbuf,
                                              hbuf, nullptr, M_, E_, E_);

        k_layernorm<<<M_, 256, 0, stream>>>(hbuf, ln2_g + (size_t)l*E_, ln2_b + (size_t)l*E_, x16);

        dim3 gf(4*E_/64, M_/16);
        k_gemm_wmma<2><<<gf, 32, 0, stream>>>(x16, wfT + (size_t)l*E_*4*E_,
                                              b_fc + (size_t)l*4*E_, nullptr,
                                              nullptr, fc16, M_, 4*E_, E_);

        dim3 gm(E_/64, M_/16);
        k_gemm_wmma<1><<<gm, 32, 0, stream>>>(fc16, wmT + (size_t)l*4*E_*E_,
                                              b_mo + (size_t)l*E_, hbuf,
                                              hbuf, nullptr, M_, E_, 4*E_);
    }

    // --- final LN + CP head GEMM
    k_layernorm<<<M_, 256, 0, stream>>>(hbuf, lnf_g, lnf_b, x16);
    dim3 gc(HCP_*V_*R_/64, M_/16);
    k_gemm_wmma<0><<<gc, 32, 0, stream>>>(x16, wcpT, b_cp, nullptr,
                                          cpbuf, nullptr, M_, HCP_*V_*R_, E_);

    // --- loss
    k_cploss<<<NVALID_, 256, 0, stream>>>(cpbuf, ids, losses);
    k_reduce_loss<<<1, 256, 0, stream>>>(losses, out);
}
